// Block_69870527971953
// MI455X (gfx1250) — compile-verified, matrix operand order checked by reading, not run on hardware
//
#include <hip/hip_runtime.h>
#include <hip/hip_bf16.h>
#include <math.h>

// ---------------- types ----------------
typedef _Float16 half_t;
typedef __attribute__((ext_vector_type(16))) _Float16 v16h;
typedef __attribute__((ext_vector_type(8)))  float    v8f;

union Frag16 { v16h v; half_t h[16]; uint4 q[2]; };
union FragC  { v8f  v; float  f[8]; };

__device__ __forceinline__ v8f wmma16(v16h a, v16h b, v8f c) {
  return __builtin_amdgcn_wmma_f32_16x16x32_f16(false, a, false, b, (short)0, c, false, false);
}

// ---------------- problem constants ----------------
#define B_   2
#define S_   1024
#define M_   576
#define D_   768
#define H_   12
#define HD_  64
#define NE_  6
#define DFF_ 3072

__device__ __forceinline__ float gelu_f(float x) {
  const float c = 0.7978845608028654f;
  float t = tanhf(c * (x + 0.044715f * x * x * x));
  return 0.5f * x * (1.0f + t);
}

// ---------------- f32 -> f16 convert ----------------
__global__ void convert_f16_kernel(const float* __restrict__ in, half_t* __restrict__ out, int n) {
  int i = blockIdx.x * blockDim.x + threadIdx.x;
  if (i < n) out[i] = (half_t)in[i];
}

// ---------------- LayerNorm (f32 rows -> f16 rows) ----------------
__device__ __forceinline__ float block_sum256(float v, float* sred) {
  #pragma unroll
  for (int m = 1; m < 32; m <<= 1) v += __shfl_xor(v, m);
  int wave = threadIdx.x >> 5, lane = threadIdx.x & 31;
  if (lane == 0) sred[wave] = v;
  __syncthreads();
  float t = (lane < 8) ? sred[lane] : 0.0f;
  #pragma unroll
  for (int m = 1; m < 8; m <<= 1) t += __shfl_xor(t, m);
  if (threadIdx.x == 0) sred[0] = t;
  __syncthreads();
  float r = sred[0];
  __syncthreads();
  return r;
}

__global__ __launch_bounds__(256) void ln_f16_kernel(const float* __restrict__ X,
                                                     const float* __restrict__ g,
                                                     const float* __restrict__ bta,
                                                     half_t* __restrict__ Y, int Dn) {
  __shared__ float sred[8];
  size_t row = blockIdx.x;
  const float* x = X + row * (size_t)Dn;
  float s = 0.0f;
  for (int i = threadIdx.x; i < Dn; i += 256) s += x[i];
  float mu = block_sum256(s, sred) / (float)Dn;
  float vv = 0.0f;
  for (int i = threadIdx.x; i < Dn; i += 256) { float d = x[i] - mu; vv += d * d; }
  float var = block_sum256(vv, sred) / (float)Dn;
  float inv = rsqrtf(var + 1e-5f);
  half_t* y = Y + row * (size_t)Dn;
  for (int i = threadIdx.x; i < Dn; i += 256)
    y[i] = (half_t)((x[i] - mu) * inv * g[i] + bta[i]);
}

// ---------------- tiled WMMA GEMM, double-buffered, templated epilogue ----------------
// C[M,N] = A[M,K] @ W[K,N]; grid = (N/128, M/128), 256 threads (8 waves, 2x4)
#define BM 128
#define BN 128
#define BK 32

template<int ACT, bool HAS_BIAS, bool HAS_RES, bool HAS_MASK, bool WF32, bool WF16>
__global__ __launch_bounds__(256) void gemm_f16_kernel(
    const half_t* __restrict__ A, const half_t* __restrict__ W,
    const float* __restrict__ bias, const float* __restrict__ res,
    const float* __restrict__ rowmask,
    float* __restrict__ Cf, half_t* __restrict__ Ch,
    int N, int K) {
  __shared__ __align__(16) half_t As[2][BM][BK + 8];
  __shared__ __align__(16) half_t Wt[2][BN][BK + 8];

  int tid  = threadIdx.x;
  int wave = tid >> 5;
  int lane = tid & 31;
  int col  = lane & 15;
  int hs   = lane >> 4;

  int bm = blockIdx.y * BM;
  int bn = blockIdx.x * BN;
  int wm = (wave >> 2) * 64;   // 0 / 64
  int wn = (wave & 3) * 32;    // 0 / 32 / 64 / 96

  // per-thread load assignments
  int ar  = tid >> 1;          // A row 0..127
  int ac  = (tid & 1) * 16;    // A k-chunk 0/16
  int wk  = tid >> 3;          // W k-row 0..31
  int wn0 = (tid & 7) * 16;    // W n-chunk 0..112
  const half_t* Aptr = A + (size_t)(bm + ar) * K + ac;
  const half_t* Wptr = W + (size_t)wk * N + bn + wn0;

  FragC acc[4][2];
  #pragma unroll
  for (int i = 0; i < 4; ++i)
    #pragma unroll
    for (int j = 0; j < 2; ++j)
      #pragma unroll
      for (int r = 0; r < 8; ++r) acc[i][j].f[r] = 0.0f;

  // preload tile 0
  {
    uint4 a0 = *(const uint4*)(Aptr);
    uint4 a1 = *(const uint4*)(Aptr + 8);
    Frag16 wu;
    wu.q[0] = *(const uint4*)(Wptr);
    wu.q[1] = *(const uint4*)(Wptr + 8);
    *(uint4*)&As[0][ar][ac]     = a0;
    *(uint4*)&As[0][ar][ac + 8] = a1;
    #pragma unroll
    for (int j = 0; j < 16; ++j) Wt[0][wn0 + j][wk] = wu.h[j];
  }
  __syncthreads();

  int nt = K / BK;
  for (int t = 0; t < nt; ++t) {
    int cur = t & 1;
    bool pf = (t + 1 < nt);
    uint4 a0, a1;
    Frag16 wu;
    if (pf) {  // register-stage next tile while WMMAs run on current
      const half_t* ap = Aptr + (size_t)(t + 1) * BK;
      a0 = *(const uint4*)ap;
      a1 = *(const uint4*)(ap + 8);
      const half_t* wp = Wptr + (size_t)(t + 1) * BK * N;
      wu.q[0] = *(const uint4*)wp;
      wu.q[1] = *(const uint4*)(wp + 8);
    }

    Frag16 afr[4], bfr[2];
    #pragma unroll
    for (int i = 0; i < 4; ++i) {
      const half_t* p = &As[cur][wm + i * 16 + col][hs * 8];
      afr[i].q[0] = *(const uint4*)p;
      afr[i].q[1] = *(const uint4*)(p + 16);
    }
    #pragma unroll
    for (int j = 0; j < 2; ++j) {
      const half_t* p = &Wt[cur][wn + j * 16 + col][hs * 8];
      bfr[j].q[0] = *(const uint4*)p;
      bfr[j].q[1] = *(const uint4*)(p + 16);
    }
    #pragma unroll
    for (int i = 0; i < 4; ++i)
      #pragma unroll
      for (int j = 0; j < 2; ++j)
        acc[i][j].v = wmma16(afr[i].v, bfr[j].v, acc[i][j].v);

    if (pf) {
      int nx = cur ^ 1;
      *(uint4*)&As[nx][ar][ac]     = a0;
      *(uint4*)&As[nx][ar][ac + 8] = a1;
      #pragma unroll
      for (int j = 0; j < 16; ++j) Wt[nx][wn0 + j][wk] = wu.h[j];
    }
    __syncthreads();
  }

  // epilogue (fully branch-free per template instantiation)
  #pragma unroll
  for (int i = 0; i < 4; ++i) {
    int rowb = bm + wm + i * 16 + 8 * hs;
    #pragma unroll
    for (int j = 0; j < 2; ++j) {
      int c = bn + wn + j * 16 + col;
      float bv = HAS_BIAS ? bias[c] : 0.0f;
      float rv[8];
      if constexpr (HAS_RES) {
        #pragma unroll
        for (int r = 0; r < 8; ++r) rv[r] = res[(size_t)(rowb + r) * N + c];
      }
      #pragma unroll
      for (int r = 0; r < 8; ++r) {
        int row = rowb + r;
        float v = acc[i][j].f[r] + bv;
        if constexpr (HAS_RES)  v += rv[r];
        if constexpr (ACT == 1) v = gelu_f(v);
        if constexpr (HAS_MASK) v *= rowmask[row];
        if constexpr (WF32) Cf[(size_t)row * N + c] = v;
        if constexpr (WF16) Ch[(size_t)row * N + c] = (half_t)v;
      }
    }
  }
}

// ---------------- flash attention (WMMA QK^T and PV), double-buffered K/V ----------------
// grid.x = S/64 query tiles, grid.y = (b*nkv + n)*H + h groups, 128 threads (4 waves)
__global__ __launch_bounds__(128) void attn_kernel(
    const half_t* __restrict__ Qb, int ldq,
    const half_t* __restrict__ Kb, int ldk,
    const half_t* __restrict__ Vb, int ldv,
    half_t* __restrict__ Ob, int ldo,
    int Skv, int nkv, int causal, float scale) {
  __shared__ __align__(16) half_t Kt[2][32][HD_ + 8];
  __shared__ __align__(16) half_t Vt[2][HD_][32 + 8];
  __shared__ __align__(16) half_t Pl[4][16][32 + 8];

  int tid  = threadIdx.x;
  int wave = tid >> 5;
  int lane = tid & 31;
  int col  = lane & 15;
  int hs   = lane >> 4;

  int g  = blockIdx.y;
  int h  = g % H_;
  int bn = g / H_;          // b*nkv + n
  int b  = bn / nkv;

  int qtile    = blockIdx.x * 64;
  int qrowbase = qtile + wave * 16;

  const half_t* Q  = Qb + (size_t)b  * S_  * ldq + h * HD_;
  const half_t* Kp = Kb + (size_t)bn * Skv * ldk + h * HD_;
  const half_t* Vp = Vb + (size_t)bn * Skv * ldv + h * HD_;
  half_t*       O  = Ob + (size_t)bn * S_  * ldo + h * HD_;

  // per-thread K/V tile load assignment
  int key = tid >> 2;          // 0..31
  int c0  = (tid & 3) * 16;    // 0..48

  // Q fragments (16 rows x 64 hd -> 2 x (16x32)), loaded once
  Frag16 qf[2];
  {
    const half_t* qp = Q + (size_t)(qrowbase + col) * ldq;
    #pragma unroll
    for (int f = 0; f < 2; ++f) {
      qf[f].q[0] = *(const uint4*)(qp + f * 32 + hs * 8);
      qf[f].q[1] = *(const uint4*)(qp + f * 32 + hs * 8 + 16);
    }
  }

  FragC o[4];
  #pragma unroll
  for (int j = 0; j < 4; ++j)
    #pragma unroll
    for (int r = 0; r < 8; ++r) o[j].f[r] = 0.0f;
  float rmax[8], rsum[8];
  #pragma unroll
  for (int r = 0; r < 8; ++r) { rmax[r] = -1e30f; rsum[r] = 0.0f; }

  int kend = causal ? min(Skv, qtile + 64) : Skv;

  // preload K/V tile 0
  {
    const half_t* ks = Kp + (size_t)key * ldk + c0;
    *(uint4*)&Kt[0][key][c0]     = *(const uint4*)ks;
    *(uint4*)&Kt[0][key][c0 + 8] = *(const uint4*)(ks + 8);
    Frag16 vu;
    const half_t* vs = Vp + (size_t)key * ldv + c0;
    vu.q[0] = *(const uint4*)vs;
    vu.q[1] = *(const uint4*)(vs + 8);
    #pragma unroll
    for (int j = 0; j < 16; ++j) Vt[0][c0 + j][key] = vu.h[j];
  }
  __syncthreads();

  for (int kt = 0, it = 0; kt < kend; kt += 32, ++it) {
    int cur = it & 1;
    bool pf = (kt + 32 < kend);
    uint4 k0, k1;
    Frag16 vu;
    if (pf) {  // register-stage next K/V tile
      const half_t* ks = Kp + (size_t)(kt + 32 + key) * ldk + c0;
      k0 = *(const uint4*)ks;
      k1 = *(const uint4*)(ks + 8);
      const half_t* vs = Vp + (size_t)(kt + 32 + key) * ldv + c0;
      vu.q[0] = *(const uint4*)vs;
      vu.q[1] = *(const uint4*)(vs + 8);
    }

    // scores: 16 q rows x 32 keys as two 16x16 C frags
    FragC sfr[2];
    #pragma unroll
    for (int g2 = 0; g2 < 2; ++g2) {
      #pragma unroll
      for (int r = 0; r < 8; ++r) sfr[g2].f[r] = 0.0f;
      #pragma unroll
      for (int f = 0; f < 2; ++f) {
        Frag16 kf;
        const half_t* p = &Kt[cur][g2 * 16 + col][f * 32 + hs * 8];
        kf.q[0] = *(const uint4*)p;
        kf.q[1] = *(const uint4*)(p + 16);
        sfr[g2].v = wmma16(qf[f].v, kf.v, sfr[g2].v);
      }
      #pragma unroll
      for (int r = 0; r < 8; ++r) {
        float v = sfr[g2].f[r] * scale;
        int kidx = kt + g2 * 16 + col;
        int qr   = qrowbase + r + 8 * hs;
        sfr[g2].f[r] = (causal && kidx > qr) ? -10000.0f : v;
      }
    }

    // row-wise streaming softmax (reductions stay inside 16-lane halves)
    float mnew[8];
    #pragma unroll
    for (int r = 0; r < 8; ++r) {
      float m = fmaxf(sfr[0].f[r], sfr[1].f[r]);
      m = fmaxf(m, __shfl_xor(m, 1));
      m = fmaxf(m, __shfl_xor(m, 2));
      m = fmaxf(m, __shfl_xor(m, 4));
      m = fmaxf(m, __shfl_xor(m, 8));
      mnew[r] = fmaxf(rmax[r], m);
    }
    #pragma unroll
    for (int r = 0; r < 8; ++r) {
      float fct = __expf(rmax[r] - mnew[r]);
      rsum[r] *= fct;
      #pragma unroll
      for (int j = 0; j < 4; ++j) o[j].f[r] *= fct;
      rmax[r] = mnew[r];
    }
    #pragma unroll
    for (int g2 = 0; g2 < 2; ++g2) {
      #pragma unroll
      for (int r = 0; r < 8; ++r) {
        float p = __expf(sfr[g2].f[r] - rmax[r]);
        float ps = p;
        ps += __shfl_xor(ps, 1);
        ps += __shfl_xor(ps, 2);
        ps += __shfl_xor(ps, 4);
        ps += __shfl_xor(ps, 8);
        rsum[r] += ps;
        Pl[wave][r + 8 * hs][g2 * 16 + col] = (half_t)p;
      }
    }
    // wave-local LDS RAW: order P stores before P loads
    asm volatile("s_wait_dscnt 0" ::: "memory");

    // P (16x32) A-fragment, then P @ V
    Frag16 pfr;
    {
      const half_t* pp = &Pl[wave][col][hs * 8];
      pfr.q[0] = *(const uint4*)pp;
      pfr.q[1] = *(const uint4*)(pp + 16);
    }
    #pragma unroll
    for (int j = 0; j < 4; ++j) {
      Frag16 vf;
      const half_t* vp = &Vt[cur][j * 16 + col][hs * 8];
      vf.q[0] = *(const uint4*)vp;
      vf.q[1] = *(const uint4*)(vp + 16);
      o[j].v = wmma16(pfr.v, vf.v, o[j].v);
    }

    if (pf) {
      int nx = cur ^ 1;
      *(uint4*)&Kt[nx][key][c0]     = k0;
      *(uint4*)&Kt[nx][key][c0 + 8] = k1;
      #pragma unroll
      for (int j = 0; j < 16; ++j) Vt[nx][c0 + j][key] = vu.h[j];
    }
    __syncthreads();
  }

  // finalize: divide by row sums, write f16 merged-head output
  #pragma unroll
  for (int r = 0; r < 8; ++r) {
    float inv = 1.0f / rsum[r];
    int qr = qrowbase + r + 8 * hs;
    #pragma unroll
    for (int j = 0; j < 4; ++j)
      O[(size_t)qr * ldo + j * 16 + col] = (half_t)(o[j].f[r] * inv);
  }
}

// ---------------- present = stack([k, v]) copy ----------------
__global__ void present_kernel(const float* __restrict__ qkv, float* __restrict__ out, int n) {
  int idx = blockIdx.x * blockDim.x + threadIdx.x;
  if (idx >= n) return;
  int hd = idx % HD_;
  int s  = (idx / HD_) % S_;
  int h  = (idx / (HD_ * S_)) % H_;
  int b  = (idx / (HD_ * S_ * H_)) % B_;
  int p  = idx / (HD_ * S_ * H_ * B_);
  out[idx] = qkv[((size_t)(b * S_ + s)) * (3 * D_) + D_ + p * D_ + h * HD_ + hd];
}

// ---------------- 6-branch cross-attn reduction ----------------
__global__ void cross_reduce_kernel(const float* __restrict__ caP,
                                    const float* __restrict__ maskq,
                                    float* __restrict__ bout, int n) {
  int idx = blockIdx.x * blockDim.x + threadIdx.x;
  if (idx >= n) return;
  int d = idx % D_;
  int s = (idx / D_) % S_;
  int b = idx / (D_ * S_);
  float acc = 0.0f;
  #pragma unroll
  for (int nn = 0; nn < NE_; ++nn)
    acc += caP[(((size_t)(b * NE_ + nn)) * S_ + s) * D_ + d];
  bout[idx] = acc * 0.40824829046386302f * maskq[b * S_ + s];
}

// ---------------- host side ----------------
extern "C" void kernel_launch(void* const* d_in, const int* in_sizes, int n_in,
                              void* d_out, int out_size, void* d_ws, size_t ws_size,
                              hipStream_t stream) {
  (void)in_sizes; (void)n_in; (void)out_size; (void)ws_size;
  const float* x      = (const float*)d_in[0];
  const float* enc    = (const float*)d_in[1];
  const float* maskq  = (const float*)d_in[2];
  // d_in[3], d_in[4]: attention masks handled analytically (causal / all-false)
  const float* ln1_g  = (const float*)d_in[5];
  const float* ln1_b  = (const float*)d_in[6];
  const float* ln2_g  = (const float*)d_in[7];
  const float* ln2_b  = (const float*)d_in[8];
  const float* attn_w = (const float*)d_in[9];
  const float* attn_b = (const float*)d_in[10];
  const float* aproj_w = (const float*)d_in[11];
  const float* aproj_b = (const float*)d_in[12];
  const float* q_w = (const float*)d_in[13];
  const float* q_b = (const float*)d_in[14];
  const float* k_w = (const float*)d_in[15];
  const float* k_b = (const float*)d_in[16];
  const float* v_w = (const float*)d_in[17];
  const float* v_b = (const float*)d_in[18];
  const float* cproj_w = (const float*)d_in[19];
  const float* cproj_b = (const float*)d_in[20];
  const float* fc_w = (const float*)d_in[21];
  const float* fc_b = (const float*)d_in[22];
  const float* mproj_w = (const float*)d_in[23];
  const float* mproj_b = (const float*)d_in[24];

  const size_t BS   = (size_t)B_ * S_;          // 2048
  const size_t BSD  = BS * D_;                  // 1572864
  const size_t BS3D = BS * 3 * D_;
  const size_t B6M  = (size_t)B_ * NE_ * M_;    // 6912
  const size_t B6MD = B6M * D_;
  const size_t B6S  = (size_t)B_ * NE_ * S_;    // 12288
  const size_t B6SD = B6S * D_;
  const size_t BSF  = BS * DFF_;

  char* ws = (char*)d_ws;
  size_t off = 0;
  auto alloc = [&](size_t bytes) -> char* {
    char* p = ws + off;
    off = (off + bytes + 255) & ~(size_t)255;
    return p;
  };

  half_t* wAttn  = (half_t*)alloc((size_t)D_ * 3 * D_ * 2);
  half_t* wAproj = (half_t*)alloc((size_t)D_ * D_ * 2);
  half_t* wQ     = (half_t*)alloc((size_t)D_ * D_ * 2);
  half_t* wK     = (half_t*)alloc((size_t)D_ * D_ * 2);
  half_t* wV     = (half_t*)alloc((size_t)D_ * D_ * 2);
  half_t* wCproj = (half_t*)alloc((size_t)D_ * D_ * 2);
  half_t* wFc    = (half_t*)alloc((size_t)D_ * DFF_ * 2);
  half_t* wMproj = (half_t*)alloc((size_t)DFF_ * D_ * 2);

  half_t* xnH  = (half_t*)alloc(BSD * 2);
  float*  qkvF = (float*) alloc(BS3D * 4);
  half_t* qkvH = (half_t*)alloc(BS3D * 2);
  half_t* saH  = (half_t*)alloc(BSD * 2);
  float*  aF   = (float*) alloc(BSD * 4);
  half_t* anH  = (half_t*)alloc(BSD * 2);
  half_t* enH  = (half_t*)alloc(B6MD * 2);
  half_t* cqH  = (half_t*)alloc(BSD * 2);
  half_t* ckH  = (half_t*)alloc(B6MD * 2);
  half_t* cvH  = (half_t*)alloc(B6MD * 2);
  half_t* caH  = (half_t*)alloc(B6SD * 2);
  float*  caPF = (float*) alloc(B6SD * 4);
  float*  bF   = (float*) alloc(BSD * 4);
  half_t* bnH  = (half_t*)alloc(BSD * 2);
  half_t* hH   = (half_t*)alloc(BSF * 2);

  auto cvt = [&](const float* src, half_t* dst, size_t n) {
    convert_f16_kernel<<<(unsigned)((n + 255) / 256), 256, 0, stream>>>(src, dst, (int)n);
  };
  cvt(attn_w,  wAttn,  (size_t)D_ * 3 * D_);
  cvt(aproj_w, wAproj, (size_t)D_ * D_);
  cvt(q_w,     wQ,     (size_t)D_ * D_);
  cvt(k_w,     wK,     (size_t)D_ * D_);
  cvt(v_w,     wV,     (size_t)D_ * D_);
  cvt(cproj_w, wCproj, (size_t)D_ * D_);
  cvt(fc_w,    wFc,    (size_t)D_ * DFF_);
  cvt(mproj_w, wMproj, (size_t)DFF_ * D_);

  // 1) xn = LN1(x) -> f16
  ln_f16_kernel<<<(unsigned)BS, 256, 0, stream>>>(x, ln1_g, ln1_b, xnH, D_);

  // 2) qkv = xn @ attn_w + attn_b (f32 + f16)
  gemm_f16_kernel<0, true, false, false, true, true>
      <<<dim3(3 * D_ / BN, (unsigned)(BS / BM)), 256, 0, stream>>>(
      xnH, wAttn, attn_b, nullptr, nullptr, qkvF, qkvH, 3 * D_, D_);

  // 3) present (k, v) to d_out tail
  {
    int n = 2 * B_ * H_ * S_ * HD_;
    present_kernel<<<(n + 255) / 256, 256, 0, stream>>>(qkvF, (float*)d_out + BSD, n);
  }

  // 4) causal self attention -> saH (merged heads, f16)
  attn_kernel<<<dim3(S_ / 64, B_ * H_), 128, 0, stream>>>(
      qkvH, 3 * D_, qkvH + D_, 3 * D_, qkvH + 2 * D_, 3 * D_,
      saH, D_, S_, 1, 1, 0.125f);

  // 5) a = x + sa @ attn_proj_w + attn_proj_b (f32)
  gemm_f16_kernel<0, true, true, false, true, false>
      <<<dim3(D_ / BN, (unsigned)(BS / BM)), 256, 0, stream>>>(
      saH, wAproj, aproj_b, x, nullptr, aF, nullptr, D_, D_);

  // 6) an = LN1(a), en = LN1(encoder_output)
  ln_f16_kernel<<<(unsigned)BS, 256, 0, stream>>>(aF, ln1_g, ln1_b, anH, D_);
  ln_f16_kernel<<<(unsigned)B6M, 256, 0, stream>>>(enc, ln1_g, ln1_b, enH, D_);

  // 7) cq / ck / cv projections (f16 only)
  gemm_f16_kernel<0, true, false, false, false, true>
      <<<dim3(D_ / BN, (unsigned)(BS / BM)), 256, 0, stream>>>(
      anH, wQ, q_b, nullptr, nullptr, nullptr, cqH, D_, D_);
  gemm_f16_kernel<0, true, false, false, false, true>
      <<<dim3(D_ / BN, (unsigned)(B6M / BM)), 256, 0, stream>>>(
      enH, wK, k_b, nullptr, nullptr, nullptr, ckH, D_, D_);
  gemm_f16_kernel<0, true, false, false, false, true>
      <<<dim3(D_ / BN, (unsigned)(B6M / BM)), 256, 0, stream>>>(
      enH, wV, v_b, nullptr, nullptr, nullptr, cvH, D_, D_);

  // 8) 6-branch cross attention -> caH
  attn_kernel<<<dim3(S_ / 64, B_ * NE_ * H_), 128, 0, stream>>>(
      cqH, D_, ckH, D_, cvH, D_, caH, D_, M_, NE_, 0, 0.125f);

  // 9) per-branch c_proj -> caPF (f32)
  gemm_f16_kernel<0, true, false, false, true, false>
      <<<dim3(D_ / BN, (unsigned)(B6S / BM)), 256, 0, stream>>>(
      caH, wCproj, cproj_b, nullptr, nullptr, caPF, nullptr, D_, D_);

  // 10) b = mask * sum_n(caP) / sqrt(6)
  cross_reduce_kernel<<<(unsigned)(BSD / 256), 256, 0, stream>>>(caPF, maskq, bF, (int)BSD);

  // 11) bn = LN2(b)
  ln_f16_kernel<<<(unsigned)BS, 256, 0, stream>>>(bF, ln2_g, ln2_b, bnH, D_);

  // 12) h = gelu(bn @ fc_w + fc_b) (f16)
  gemm_f16_kernel<1, true, false, false, false, true>
      <<<dim3(DFF_ / BN, (unsigned)(BS / BM)), 256, 0, stream>>>(
      bnH, wFc, fc_b, nullptr, nullptr, nullptr, hH, DFF_, D_);

  // 13) out = (a + h @ mlp_proj_w + mlp_proj_b) * mask_queries (f32 -> d_out)
  gemm_f16_kernel<0, true, true, true, true, false>
      <<<dim3(D_ / BN, (unsigned)(BS / BM)), 256, 0, stream>>>(
      hH, wMproj, mproj_b, aF, maskq, (float*)d_out, nullptr, D_, DFF_);
}
